// SelectivePSI_49950469653073
// MI455X (gfx1250) — matrix operand order repaired and checked by reading, not compiled, not run on hardware
//
#include <hip/hip_runtime.h>
#include <hip/hip_bf16.h>
#include <math.h>

// problem dims
#define BB 4
#define SS 4096
#define DD 1024
#define RR 128
#define MM (BB*SS)        // 16384 rows
#define TWO_D (2*DD)
#define NCH 16            // scan chunks per channel
#define CHL (SS/NCH)      // 256 steps per chunk

typedef __attribute__((ext_vector_type(8)))  __bf16 v8bf;
typedef __attribute__((ext_vector_type(16))) __bf16 v16bf;
typedef __attribute__((ext_vector_type(8)))  float  v8f;

enum { EPI_NONE=0, EPI_BIAS, EPI_BIAS2, EPI_SIGMOID, EPI_SIGMOID5, EPI_SOFTPLUS, EPI_GELU, EPI_RESID };

template<int EPI>
__device__ __forceinline__ float apply_epi(float v, int n, size_t o,
    const float* __restrict__ b1, const float* __restrict__ b2,
    const float* __restrict__ resid)
{
  if constexpr (EPI == EPI_NONE) { (void)n; (void)o; (void)b1; (void)b2; (void)resid; return v; }
  else {
    v += b1[n];
    if constexpr (EPI == EPI_BIAS2)    v += b2[n];
    if constexpr (EPI == EPI_SIGMOID)  v = 1.0f/(1.0f+expf(-v));
    if constexpr (EPI == EPI_SIGMOID5) v = 5.0f/(1.0f+expf(-v));
    if constexpr (EPI == EPI_SOFTPLUS) v = (v > 20.0f) ? v : log1pf(expf(v));
    if constexpr (EPI == EPI_GELU)     v = 0.5f*v*(1.0f+erff(v*0.70710678118654752f));
    if constexpr (EPI == EPI_RESID)    v += resid[o];
    return v;
  }
}

// A-fragment (ISA 7.12.2, 16-bit A 16x32): lane (g,m): halves j<8 -> K=k+8g+j;
// j>=8 -> K=k+16+8g+(j-8)  => two b128 loads at +0 / +16 (pointer pre-offset by 8g).
__device__ __forceinline__ v16bf load_afrag(const __bf16* p) {
  v8bf a0 = *(const v8bf*)(p);
  v8bf a1 = *(const v8bf*)(p + 16);
  return __builtin_shufflevector(a0, a1, 0,1,2,3,4,5,6,7,8,9,10,11,12,13,14,15);
}
// B-fragment: lane (g,n): halves j<8 -> K=k+8g+j; j>=8 -> K=k+16+8g+(j-8),
// elem B[K][n]=Bt[n][K] => two b128 loads at +0 / +16 from row n of Bt (pointer pre-offset by 8g).
__device__ __forceinline__ v16bf load_bfrag(const __bf16* p) {
  v8bf b0 = *(const v8bf*)(p);
  v8bf b1 = *(const v8bf*)(p + 16);
  return __builtin_shufflevector(b0, b1, 0,1,2,3,4,5,6,7,8,9,10,11,12,13,14,15);
}

// ---------------------------------------------------------------------------
// bf16 WMMA GEMM:  C[M,N] = A[M,K] @ B[K,N], B pre-transposed as Bt[N,K].
// Wave32: each wave computes a 32x64 tile: 2 M-subtiles x 4 N-subtiles
// (8x v_wmma_f32_16x16x32_bf16 per 32-wide K-step, B frags shared by both
// M-subtiles -> 1.5 b128 loads per WMMA). 8 waves/block -> 32x512 per block.
// K-loop unrolled by 64 with ping-pong fragment buffers so next-step loads
// overlap current-step WMMAs. Requires: M%32==0, N%64==0, K%64==0.
// C/D layout: VGPR r holds row m0 + r + 8g, col n0 + (lane&15).
// ---------------------------------------------------------------------------
template<int EPI, bool SBF16>
__global__ __launch_bounds__(256)
void gemm_bf16_wmma(const __bf16* __restrict__ A, const __bf16* __restrict__ Bt,
                    void* __restrict__ Cout,
                    const float* __restrict__ bias1, const float* __restrict__ bias2,
                    const float* __restrict__ resid,
                    int M, int N, int K)
{
  const int wave = threadIdx.x >> 5;
  const int lane = threadIdx.x & 31;
  const int g    = lane >> 4;
  const int mn   = lane & 15;
  const int m0   = blockIdx.x * 32;
  const int n_base = blockIdx.y * 512 + wave * 64;
  if (n_base + 64 > N) return;          // N is a multiple of 64 in all uses
  (void)M;

  v8f acc[2][4];
  #pragma unroll
  for (int u = 0; u < 2; ++u)
    #pragma unroll
    for (int t = 0; t < 4; ++t)
      #pragma unroll
      for (int e = 0; e < 8; ++e) acc[u][t][e] = 0.0f;

  const __bf16* __restrict__ arow0 = A + (size_t)(m0 + mn) * K + 8*g;
  const __bf16* __restrict__ arow1 = arow0 + (size_t)16 * K;
  const __bf16* brow[4];
  #pragma unroll
  for (int t = 0; t < 4; ++t)
    brow[t] = Bt + (size_t)(n_base + 16*t + mn) * K + 8*g;

  v16bf a_cur[2], b_cur[4], a_nxt[2], b_nxt[4];
  a_cur[0] = load_afrag(arow0);
  a_cur[1] = load_afrag(arow1);
  #pragma unroll
  for (int t = 0; t < 4; ++t) b_cur[t] = load_bfrag(brow[t]);

  for (int k0 = 0; k0 < K; k0 += 64) {
    // prefetch k0+32 (always valid: K % 64 == 0)
    a_nxt[0] = load_afrag(arow0 + k0 + 32);
    a_nxt[1] = load_afrag(arow1 + k0 + 32);
    #pragma unroll
    for (int t = 0; t < 4; ++t) b_nxt[t] = load_bfrag(brow[t] + k0 + 32);

    #pragma unroll
    for (int t = 0; t < 4; ++t) {
      acc[0][t] = __builtin_amdgcn_wmma_f32_16x16x32_bf16(
                    false, a_cur[0], false, b_cur[t], (short)0, acc[0][t], false, false);
      acc[1][t] = __builtin_amdgcn_wmma_f32_16x16x32_bf16(
                    false, a_cur[1], false, b_cur[t], (short)0, acc[1][t], false, false);
    }

    if (k0 + 64 < K) {          // prefetch k0+64 for next iteration
      a_cur[0] = load_afrag(arow0 + k0 + 64);
      a_cur[1] = load_afrag(arow1 + k0 + 64);
      #pragma unroll
      for (int t = 0; t < 4; ++t) b_cur[t] = load_bfrag(brow[t] + k0 + 64);
    }

    #pragma unroll
    for (int t = 0; t < 4; ++t) {
      acc[0][t] = __builtin_amdgcn_wmma_f32_16x16x32_bf16(
                    false, a_nxt[0], false, b_nxt[t], (short)0, acc[0][t], false, false);
      acc[1][t] = __builtin_amdgcn_wmma_f32_16x16x32_bf16(
                    false, a_nxt[1], false, b_nxt[t], (short)0, acc[1][t], false, false);
    }
  }

  #pragma unroll
  for (int u = 0; u < 2; ++u) {
    #pragma unroll
    for (int t = 0; t < 4; ++t) {
      const int n = n_base + 16*t + mn;
      #pragma unroll
      for (int r = 0; r < 8; ++r) {
        const int row = m0 + 16*u + r + 8*g;
        const size_t o = (size_t)row * N + n;
        float v = apply_epi<EPI>(acc[u][t][r], n, o, bias1, bias2, resid);
        if constexpr (SBF16) ((__bf16*)Cout)[o] = (__bf16)v;
        else                 ((float*) Cout)[o] = v;
      }
    }
  }
}

// ---------------------------------------------------------------------------
// packing kernels
// ---------------------------------------------------------------------------
__global__ void cast_bf16_k(const float* __restrict__ in, __bf16* __restrict__ out, int n) {
  int i = blockIdx.x * blockDim.x + threadIdx.x;
  if (i < n) out[i] = (__bf16)in[i];
}

// W[K,N] row-major -> Wt[N,K] bf16
__global__ void packT_k(const float* __restrict__ W, __bf16* __restrict__ Wt, int K, int N) {
  int i = blockIdx.x * blockDim.x + threadIdx.x;
  if (i < K * N) {
    int k = i / N, n = i % N;
    Wt[(size_t)n * K + k] = (__bf16)W[(size_t)k * N + n];
  }
}

// ---------------------------------------------------------------------------
// chunked scan over S (all scan operators are additions; phi prefix enters
// cos/sin only pointwise, so chunk-sum -> prefix -> replay is exact).
// channel c = b*DD + d ; chunk j ; tid = j*(BB*DD) + c (coalesced over d).
// ---------------------------------------------------------------------------
__global__ void scan_pass1(const float* __restrict__ omega, const float* __restrict__ dt,
                           float* __restrict__ chunkO)
{
  int tid = blockIdx.x * blockDim.x + threadIdx.x;          // < BB*DD*NCH
  int j = tid / (BB*DD);
  int c = tid % (BB*DD);
  int b = c / DD, d = c % DD;
  size_t base = ((size_t)b * SS + (size_t)j * CHL) * DD + d;
  float s = 0.f;
  for (int i = 0; i < CHL; ++i) {
    size_t idx = base + (size_t)i * DD;
    s += omega[idx] * dt[idx] * 0.01f;
  }
  chunkO[c * NCH + j] = s;
}

__global__ void scan_prefix(float* __restrict__ ch) {   // in-place exclusive prefix
  int c = blockIdx.x * blockDim.x + threadIdx.x;        // < BB*DD
  float run = 0.f;
  #pragma unroll
  for (int j = 0; j < NCH; ++j) {
    float t = ch[c * NCH + j];
    ch[c * NCH + j] = run;
    run += t;
  }
}

__global__ void scan_pass2(const float* __restrict__ omega, const float* __restrict__ dt,
                           const float* __restrict__ phi_init, const float* __restrict__ gate,
                           const float* __restrict__ x, const float* __restrict__ mag,
                           const float* __restrict__ chunkO,
                           float* __restrict__ chunkR, float* __restrict__ chunkI,
                           float* __restrict__ chunkA)
{
  int tid = blockIdx.x * blockDim.x + threadIdx.x;
  int j = tid / (BB*DD);
  int c = tid % (BB*DD);
  int b = c / DD, d = c % DD;
  size_t base = ((size_t)b * SS + (size_t)j * CHL) * DD + d;
  float phi_run = chunkO[c * NCH + j];     // exclusive prefix of omega_scaled
  float sr = 0.f, si = 0.f, sa = 0.f;
  for (int i = 0; i < CHL; ++i) {
    size_t idx = base + (size_t)i * DD;
    float os = omega[idx] * dt[idx] * 0.01f;
    phi_run += os;
    float phi = phi_init[idx] + phi_run;
    float gt = gate[idx], mg = mag[idx];
    float ct = gt * x[idx] * mg;
    sr += ct * cosf(phi);
    si += ct * sinf(phi);
    sa += mg * gt;
  }
  chunkR[c * NCH + j] = sr;
  chunkI[c * NCH + j] = si;
  chunkA[c * NCH + j] = sa;
}

__global__ void scan_pass3(const float* __restrict__ omega, const float* __restrict__ dt,
                           const float* __restrict__ phi_init, const float* __restrict__ gate,
                           const float* __restrict__ x, const float* __restrict__ mag,
                           const float* __restrict__ qoff,
                           const float* __restrict__ chunkO, const float* __restrict__ chunkR,
                           const float* __restrict__ chunkI, const float* __restrict__ chunkA,
                           float* __restrict__ ctx)
{
  int tid = blockIdx.x * blockDim.x + threadIdx.x;
  int j = tid / (BB*DD);
  int c = tid % (BB*DD);
  int b = c / DD, d = c % DD;
  size_t base = ((size_t)b * SS + (size_t)j * CHL) * DD + d;
  int cj = c * NCH + j;
  float phi_run = chunkO[cj];
  float mr = chunkR[cj], mi = chunkI[cj], ma = chunkA[cj];
  for (int i = 0; i < CHL; ++i) {
    size_t idx = base + (size_t)i * DD;
    float os = omega[idx] * dt[idx] * 0.01f;
    phi_run += os;
    float phi = phi_init[idx] + phi_run;
    float gt = gate[idx], mg = mag[idx];
    float ct = gt * x[idx] * mg;
    mr += ct * cosf(phi);
    mi += ct * sinf(phi);
    ma += mg * gt;
    float sm  = sqrtf(ma + 1e-8f);
    float mrn = mr / sm, min_ = mi / sm;
    float pq = phi + qoff[idx];
    float cq = cosf(pq), sq = sinf(pq);
    float rr = mrn * cq + min_ * sq;
    float ri = min_ * cq - mrn * sq;
    int srow = b * SS + j * CHL + i;
    ctx[(size_t)srow * TWO_D + d]      = rr;
    ctx[(size_t)srow * TWO_D + DD + d] = ri;
  }
}

// ---------------------------------------------------------------------------
// LayerNorm over last dim (2048), writes bf16 for the following WMMA GEMM
// ---------------------------------------------------------------------------
__global__ __launch_bounds__(256)
void layernorm_k(const float* __restrict__ ctx, const float* __restrict__ gam,
                 const float* __restrict__ bet, __bf16* __restrict__ out)
{
  __shared__ float s1[256], s2[256];
  const int row = blockIdx.x, t = threadIdx.x;
  const float* rp = ctx + (size_t)row * TWO_D;
  float a = 0.f, b = 0.f;
  #pragma unroll
  for (int i = 0; i < TWO_D/256; ++i) {
    float v = rp[t + i*256];
    a += v; b += v*v;
  }
  s1[t] = a; s2[t] = b;
  __syncthreads();
  for (int st = 128; st > 0; st >>= 1) {
    if (t < st) { s1[t] += s1[t+st]; s2[t] += s2[t+st]; }
    __syncthreads();
  }
  const float mu   = s1[0] * (1.0f / TWO_D);
  const float var  = s2[0] * (1.0f / TWO_D) - mu*mu;
  const float rstd = rsqrtf(var + 1e-5f);
  __bf16* op = out + (size_t)row * TWO_D;
  #pragma unroll
  for (int i = 0; i < TWO_D/256; ++i) {
    int cc = t + i*256;
    op[cc] = (__bf16)((rp[cc] - mu) * rstd * gam[cc] + bet[cc]);
  }
}

// ---------------------------------------------------------------------------
extern "C" void kernel_launch(void* const* d_in, const int* in_sizes, int n_in,
                              void* d_out, int out_size, void* d_ws, size_t ws_size,
                              hipStream_t stream)
{
  (void)in_sizes; (void)n_in; (void)out_size; (void)ws_size;
  const float* x       = (const float*)d_in[0];
  const float* Wdd     = (const float*)d_in[1];   // (D,R)
  const float* Wdu     = (const float*)d_in[2];   // (R,D)
  const float* b_dtup  = (const float*)d_in[3];
  const float* Wg      = (const float*)d_in[4];
  const float* b_g     = (const float*)d_in[5];
  const float* Wp1     = (const float*)d_in[6];
  const float* b_p1    = (const float*)d_in[7];
  const float* Wp2     = (const float*)d_in[8];
  const float* b_p2    = (const float*)d_in[9];
  const float* om_base = (const float*)d_in[10];
  const float* Wom     = (const float*)d_in[11];
  const float* b_om    = (const float*)d_in[12];
  const float* Wmg     = (const float*)d_in[13];
  const float* b_mg    = (const float*)d_in[14];
  const float* Wq      = (const float*)d_in[15];
  const float* b_q     = (const float*)d_in[16];
  const float* ln_g    = (const float*)d_in[17];
  const float* ln_b    = (const float*)d_in[18];
  const float* Wo1     = (const float*)d_in[19];  // (2D,D)
  const float* b_o1    = (const float*)d_in[20];
  const float* Wo2     = (const float*)d_in[21];
  const float* b_o2    = (const float*)d_in[22];
  float* out = (float*)d_out;

  // bump allocator over workspace
  char* ws = (char*)d_ws;
  size_t off = 0;
  auto alloc = [&](size_t bytes) -> void* {
    void* p = ws + off;
    off += (bytes + 255) & ~(size_t)255;
    return p;
  };
  __bf16* xb     = (__bf16*)alloc((size_t)MM * DD * 2);
  __bf16* wddT   = (__bf16*)alloc((size_t)RR * DD * 2);
  __bf16* wduT   = (__bf16*)alloc((size_t)DD * RR * 2);
  __bf16* wgT    = (__bf16*)alloc((size_t)DD * DD * 2);
  __bf16* womT   = (__bf16*)alloc((size_t)DD * DD * 2);
  __bf16* wp1T   = (__bf16*)alloc((size_t)DD * DD * 2);
  __bf16* wp2T   = (__bf16*)alloc((size_t)DD * DD * 2);
  __bf16* wmgT   = (__bf16*)alloc((size_t)DD * DD * 2);
  __bf16* wqT    = (__bf16*)alloc((size_t)DD * DD * 2);
  __bf16* wo1T   = (__bf16*)alloc((size_t)DD * TWO_D * 2);
  __bf16* wo2T   = (__bf16*)alloc((size_t)DD * DD * 2);
  __bf16* tmp_r  = (__bf16*)alloc((size_t)MM * RR * 2);   // x @ Wdd
  __bf16* g1     = (__bf16*)alloc((size_t)MM * DD * 2);   // gelu(x@Wp1); reused as gelu(out1)
  float*  dt     = (float*)alloc((size_t)MM * DD * 4);
  float*  gate   = (float*)alloc((size_t)MM * DD * 4);
  float*  omega  = (float*)alloc((size_t)MM * DD * 4);
  float*  phi_i  = (float*)alloc((size_t)MM * DD * 4);
  float*  mag    = (float*)alloc((size_t)MM * DD * 4);
  float*  qoff   = (float*)alloc((size_t)MM * DD * 4);
  float*  ctx    = (float*)alloc((size_t)MM * TWO_D * 4);
  __bf16* lnb    = (__bf16*)alloc((size_t)MM * TWO_D * 2);
  float*  chO    = (float*)alloc((size_t)BB * DD * NCH * 4);
  float*  chR    = (float*)alloc((size_t)BB * DD * NCH * 4);
  float*  chI    = (float*)alloc((size_t)BB * DD * NCH * 4);
  float*  chA    = (float*)alloc((size_t)BB * DD * NCH * 4);

  const dim3 blk(256);
  auto cdiv = [](int a, int b) { return (a + b - 1) / b; };

  // ---- pack activations + weights to bf16 (Bt = W^T, [N,K]) ----
  cast_bf16_k<<<cdiv(MM*DD, 256), blk, 0, stream>>>(x, xb, MM*DD);
  packT_k<<<cdiv(DD*RR, 256),    blk, 0, stream>>>(Wdd, wddT, DD,    RR);
  packT_k<<<cdiv(RR*DD, 256),    blk, 0, stream>>>(Wdu, wduT, RR,    DD);
  packT_k<<<cdiv(DD*DD, 256),    blk, 0, stream>>>(Wg,  wgT,  DD,    DD);
  packT_k<<<cdiv(DD*DD, 256),    blk, 0, stream>>>(Wom, womT, DD,    DD);
  packT_k<<<cdiv(DD*DD, 256),    blk, 0, stream>>>(Wp1, wp1T, DD,    DD);
  packT_k<<<cdiv(DD*DD, 256),    blk, 0, stream>>>(Wp2, wp2T, DD,    DD);
  packT_k<<<cdiv(DD*DD, 256),    blk, 0, stream>>>(Wmg, wmgT, DD,    DD);
  packT_k<<<cdiv(DD*DD, 256),    blk, 0, stream>>>(Wq,  wqT,  DD,    DD);
  packT_k<<<cdiv(TWO_D*DD, 256), blk, 0, stream>>>(Wo1, wo1T, TWO_D, DD);
  packT_k<<<cdiv(DD*DD, 256),    blk, 0, stream>>>(Wo2, wo2T, DD,    DD);

  auto ggrid = [&](int N) { return dim3(MM / 32, cdiv(N, 512)); };

  // ---- projection GEMMs (all WMMA bf16 -> f32 acc, fused epilogues) ----
  gemm_bf16_wmma<EPI_NONE,     true ><<<ggrid(RR), blk, 0, stream>>>(xb,    wddT, tmp_r, nullptr, nullptr, nullptr, MM, RR, DD);
  gemm_bf16_wmma<EPI_SOFTPLUS, false><<<ggrid(DD), blk, 0, stream>>>(tmp_r, wduT, dt,    b_dtup,  nullptr, nullptr, MM, DD, RR);
  gemm_bf16_wmma<EPI_SIGMOID,  false><<<ggrid(DD), blk, 0, stream>>>(xb,    wgT,  gate,  b_g,     nullptr, nullptr, MM, DD, DD);
  gemm_bf16_wmma<EPI_BIAS2,    false><<<ggrid(DD), blk, 0, stream>>>(xb,    womT, omega, b_om,    om_base, nullptr, MM, DD, DD);
  gemm_bf16_wmma<EPI_GELU,     true ><<<ggrid(DD), blk, 0, stream>>>(xb,    wp1T, g1,    b_p1,    nullptr, nullptr, MM, DD, DD);
  gemm_bf16_wmma<EPI_BIAS,     false><<<ggrid(DD), blk, 0, stream>>>(g1,    wp2T, phi_i, b_p2,    nullptr, nullptr, MM, DD, DD);
  gemm_bf16_wmma<EPI_SIGMOID5, false><<<ggrid(DD), blk, 0, stream>>>(xb,    wmgT, mag,   b_mg,    nullptr, nullptr, MM, DD, DD);
  gemm_bf16_wmma<EPI_BIAS,     false><<<ggrid(DD), blk, 0, stream>>>(xb,    wqT,  qoff,  b_q,     nullptr, nullptr, MM, DD, DD);

  // ---- chunked sequential scans (exact; addition-only combine) ----
  const int scan_threads = BB * DD * NCH;           // 65536
  scan_pass1<<<cdiv(scan_threads, 256), blk, 0, stream>>>(omega, dt, chO);
  scan_prefix<<<cdiv(BB*DD, 256), blk, 0, stream>>>(chO);
  scan_pass2<<<cdiv(scan_threads, 256), blk, 0, stream>>>(omega, dt, phi_i, gate, x, mag, chO, chR, chI, chA);
  scan_prefix<<<cdiv(BB*DD, 256), blk, 0, stream>>>(chR);
  scan_prefix<<<cdiv(BB*DD, 256), blk, 0, stream>>>(chI);
  scan_prefix<<<cdiv(BB*DD, 256), blk, 0, stream>>>(chA);
  scan_pass3<<<cdiv(scan_threads, 256), blk, 0, stream>>>(omega, dt, phi_i, gate, x, mag, qoff,
                                                          chO, chR, chI, chA, ctx);

  // ---- layernorm -> bf16 ----
  layernorm_k<<<MM, blk, 0, stream>>>(ctx, ln_g, ln_b, lnb);

  // ---- output MLP + residual ----
  gemm_bf16_wmma<EPI_GELU,  true ><<<ggrid(DD), blk, 0, stream>>>(lnb, wo1T, g1,  b_o1, nullptr, nullptr, MM, DD, TWO_D);
  gemm_bf16_wmma<EPI_RESID, false><<<ggrid(DD), blk, 0, stream>>>(g1,  wo2T, out, b_o2, nullptr, x,       MM, DD, DD);
}
